// OptimizedS6Block_30571577213789
// MI455X (gfx1250) — compile-verified, matrix operand order checked by reading, not run on hardware
//
#include <hip/hip_runtime.h>
#include <hip/hip_bf16.h>

// ---------------------------------------------------------------------------
// S6 / Mamba block forward for MI455X (gfx1250, wave32, WMMA).
// GEMMs: v_wmma_f32_16x16x32_f16 with 2x2 register blocking per wave
// (32x32 output tile, 4 WMMAs per K-step from 2 A + 2 B fragments ->
// 16 flops per L2 byte). Weights pre-packed transposed-f16 so every
// fragment is two contiguous 16B global_load_b128 per lane.
// ---------------------------------------------------------------------------

typedef __attribute__((ext_vector_type(8)))  _Float16 v8h;
typedef __attribute__((ext_vector_type(16))) _Float16 v16h;
typedef __attribute__((ext_vector_type(8)))  float    v8f;

#define DMODEL 768
#define DINNER 1536
#define DSTATE 64
#define XDBL   1664            // 2*DSTATE + DINNER
#define BATCH  2
#define SEQ    1024
#define MROWS  (BATCH*SEQ)     // 2048

__device__ __forceinline__ float silu_f(float v) {
    return v / (1.0f + __expf(-v));
}

// Load one 16x32 f16 fragment (per-lane: 8 halves at koff, 8 at koff+16).
__device__ __forceinline__ v16h ld_frag(const _Float16* __restrict__ p) {
    v8h lo = *(const v8h*)p;
    v8h hi = *(const v8h*)(p + 16);
    v16h f;
#pragma unroll
    for (int i = 0; i < 8; ++i) { f[i] = lo[i]; f[8 + i] = hi[i]; }
    return f;
}

__device__ __forceinline__ v8f wmma_step(v16h a, v16h b, v8f c) {
    return __builtin_amdgcn_wmma_f32_16x16x32_f16(
        /*neg_a=*/false, a, /*neg_b=*/false, b,
        /*c_mod=*/(short)0, c, /*reuse_a=*/false, /*reuse_b=*/false);
}

struct Acc4 { v8f c[4]; };   // [im*2+in] sub-tiles of the 32x32 block

// ---------------------------------------------------------------------------
// Wave-level 32x32 tile: A MxK row-major f16, Bt NxK row-major f16 (B^T).
// 2 A-fragments x 2 B-fragments -> 4 WMMAs per 32-deep K step.
// ---------------------------------------------------------------------------
__device__ __forceinline__ Acc4 wmma_tile_2x2(const _Float16* __restrict__ A,
                                              const _Float16* __restrict__ Bt,
                                              int m0, int n0, int K, int lane) {
    const int r    = lane & 15;
    const int koff = (lane >> 4) << 3;          // 0 or 8
    const _Float16* a0p = A  + (size_t)(m0 + r)      * K + koff;
    const _Float16* a1p = A  + (size_t)(m0 + 16 + r) * K + koff;
    const _Float16* b0p = Bt + (size_t)(n0 + r)      * K + koff;
    const _Float16* b1p = Bt + (size_t)(n0 + 16 + r) * K + koff;
    Acc4 acc;
    acc.c[0] = (v8f){}; acc.c[1] = (v8f){}; acc.c[2] = (v8f){}; acc.c[3] = (v8f){};
    for (int k = 0; k < K; k += 32) {
        if (k + 96 < K) {                        // gfx1250 global_prefetch_b8
            __builtin_prefetch(a0p + k + 96, 0, 1);
            __builtin_prefetch(a1p + k + 96, 0, 1);
            __builtin_prefetch(b0p + k + 96, 0, 1);
            __builtin_prefetch(b1p + k + 96, 0, 1);
        }
        v16h a0 = ld_frag(a0p + k);
        v16h a1 = ld_frag(a1p + k);
        v16h b0 = ld_frag(b0p + k);
        v16h b1 = ld_frag(b1p + k);
        acc.c[0] = wmma_step(a0, b0, acc.c[0]);
        acc.c[1] = wmma_step(a0, b1, acc.c[1]);
        acc.c[2] = wmma_step(a1, b0, acc.c[2]);
        acc.c[3] = wmma_step(a1, b1, acc.c[3]);
    }
    return acc;
}

// ---------------------------------------------------------------------------
// Packing kernels
// ---------------------------------------------------------------------------
__global__ void k_pack_f16(const float* __restrict__ src, _Float16* __restrict__ dst,
                           int n) {
    int i = blockIdx.x * blockDim.x + threadIdx.x;
    if (i < n) dst[i] = (_Float16)src[i];
}

// dst[n*K + k] = src[k*N + n]  (f32 KxN -> f16 NxK)
__global__ void k_pack_t_f16(const float* __restrict__ src, _Float16* __restrict__ dst,
                             int K, int N) {
    int i = blockIdx.x * blockDim.x + threadIdx.x;
    if (i < N * K) {
        int n = i / K, k = i % K;
        dst[i] = (_Float16)src[(size_t)k * N + n];
    }
}

// ---------------------------------------------------------------------------
// GEMM 1: xr = X(2048x768) @ W_in(768x3072); epilogue SiLU, split halves.
// ---------------------------------------------------------------------------
__global__ void k_gemm_in(const _Float16* __restrict__ Xh,
                          const _Float16* __restrict__ WinT,
                          _Float16* __restrict__ xs_h,
                          float* __restrict__ res_silu) {
    const int N = 2 * DINNER, K = DMODEL, TN = N / 32;
    int wave = (blockIdx.x * blockDim.x + threadIdx.x) >> 5;
    int lane = threadIdx.x & 31;
    int tm = wave / TN, tn = wave % TN;
    if (tm >= MROWS / 32) return;                     // wave-uniform
    Acc4 acc = wmma_tile_2x2(Xh, WinT, tm * 32, tn * 32, K, lane);
#pragma unroll
    for (int s = 0; s < 4; ++s) {
        int n  = tn * 32 + (s & 1) * 16 + (lane & 15);
        int mb = tm * 32 + (s >> 1) * 16 + ((lane >> 4) << 3);
#pragma unroll
        for (int i = 0; i < 8; ++i) {
            float sv = silu_f(acc.c[s][i]);
            int m = mb + i;
            if (n < DINNER) xs_h[(size_t)m * DINNER + n] = (_Float16)sv;
            else            res_silu[(size_t)m * DINNER + (n - DINNER)] = sv;
        }
    }
}

// ---------------------------------------------------------------------------
// GEMM 2: x_dbl = xs(2048x1536) @ W_x(1536x1664); delta_in slice also as f16.
// ---------------------------------------------------------------------------
__global__ void k_gemm_xdbl(const _Float16* __restrict__ xs_h,
                            const _Float16* __restrict__ WxT,
                            float* __restrict__ xdbl,
                            _Float16* __restrict__ dtin_h) {
    const int N = XDBL, K = DINNER, TN = N / 32;
    int wave = (blockIdx.x * blockDim.x + threadIdx.x) >> 5;
    int lane = threadIdx.x & 31;
    int tm = wave / TN, tn = wave % TN;
    if (tm >= MROWS / 32) return;
    Acc4 acc = wmma_tile_2x2(xs_h, WxT, tm * 32, tn * 32, K, lane);
#pragma unroll
    for (int s = 0; s < 4; ++s) {
        int n  = tn * 32 + (s & 1) * 16 + (lane & 15);
        int mb = tm * 32 + (s >> 1) * 16 + ((lane >> 4) << 3);
#pragma unroll
        for (int i = 0; i < 8; ++i) {
            float v = acc.c[s][i];
            int m = mb + i;
            xdbl[(size_t)m * XDBL + n] = v;
            if (n >= DSTATE && n < 2 * DSTATE)
                dtin_h[(size_t)m * DSTATE + (n - DSTATE)] = (_Float16)v;
        }
    }
}

// ---------------------------------------------------------------------------
// GEMM 3: delta = min(softplus(dtin(2048x64) @ W_dt(64x1536) + b_dt), 10)
// ---------------------------------------------------------------------------
__global__ void k_gemm_delta(const _Float16* __restrict__ dtin_h,
                             const _Float16* __restrict__ WdtT,
                             const float* __restrict__ b_dt,
                             float* __restrict__ delta) {
    const int N = DINNER, K = DSTATE, TN = N / 32;
    int wave = (blockIdx.x * blockDim.x + threadIdx.x) >> 5;
    int lane = threadIdx.x & 31;
    int tm = wave / TN, tn = wave % TN;
    if (tm >= MROWS / 32) return;
    Acc4 acc = wmma_tile_2x2(dtin_h, WdtT, tm * 32, tn * 32, K, lane);
#pragma unroll
    for (int s = 0; s < 4; ++s) {
        int n  = tn * 32 + (s & 1) * 16 + (lane & 15);
        int mb = tm * 32 + (s >> 1) * 16 + ((lane >> 4) << 3);
        float bias = b_dt[n];
#pragma unroll
        for (int i = 0; i < 8; ++i) {
            float v = acc.c[s][i] + bias;
            float sp = (v > 20.0f) ? v : log1pf(__expf(v));   // softplus
            delta[(size_t)(mb + i) * DINNER + n] = fminf(sp, 10.0f);
        }
    }
}

// ---------------------------------------------------------------------------
// Selective scan: thread per (b,d) channel; 64 states in registers.
// Fused epilogue: y = (C*sum(h) + xs*Dp) * silu(res) -> f16.
// ---------------------------------------------------------------------------
__global__ void k_scan(const float* __restrict__ xdbl,
                       const _Float16* __restrict__ xs_h,
                       const float* __restrict__ delta,
                       const float* __restrict__ A_log,
                       const float* __restrict__ Dp,
                       const float* __restrict__ res_silu,
                       _Float16* __restrict__ y_h) {
    const int CPB = 256;                    // channels per block
    int tid = threadIdx.x;
    int b   = blockIdx.x / (DINNER / CPB);
    int d   = (blockIdx.x % (DINNER / CPB)) * CPB + tid;

    __shared__ float sB[DSTATE];
    float A[DSTATE], h[DSTATE];
#pragma unroll
    for (int n = 0; n < DSTATE; ++n) {
        A[n] = -__expf(A_log[(size_t)d * DSTATE + n]);
        h[n] = 0.0f;
    }
    float dpv = Dp[d];

    for (int t = 0; t < SEQ; ++t) {
        size_t row = (size_t)(b * SEQ + t);
        if (tid < DSTATE) sB[tid] = xdbl[row * XDBL + tid];      // B_t
        __syncthreads();
        float dt = delta[row * DINNER + d];
        float xv = (float)xs_h[row * DINNER + d];
        float cv = xdbl[row * XDBL + 2 * DSTATE + d];            // C_t
        float dx = dt * xv;
        float sum = 0.0f;
#pragma unroll
        for (int n = 0; n < DSTATE; ++n) {
            h[n] = __expf(dt * A[n]) * h[n] + dx * sB[n];
            sum += h[n];
        }
        float y = cv * sum + xv * dpv;
        y *= res_silu[row * DINNER + d];
        y_h[row * DINNER + d] = (_Float16)y;
        __syncthreads();
    }
}

// ---------------------------------------------------------------------------
// GEMM 4: out_pre = y(2048x1536) @ W_out(1536x768) + residual x
// ---------------------------------------------------------------------------
__global__ void k_gemm_out(const _Float16* __restrict__ y_h,
                           const _Float16* __restrict__ WoutT,
                           const float* __restrict__ x_res,
                           float* __restrict__ out_pre) {
    const int N = DMODEL, K = DINNER, TN = N / 32;
    int wave = (blockIdx.x * blockDim.x + threadIdx.x) >> 5;
    int lane = threadIdx.x & 31;
    int tm = wave / TN, tn = wave % TN;
    if (tm >= MROWS / 32) return;
    Acc4 acc = wmma_tile_2x2(y_h, WoutT, tm * 32, tn * 32, K, lane);
#pragma unroll
    for (int s = 0; s < 4; ++s) {
        int n  = tn * 32 + (s & 1) * 16 + (lane & 15);
        int mb = tm * 32 + (s >> 1) * 16 + ((lane >> 4) << 3);
#pragma unroll
        for (int i = 0; i < 8; ++i) {
            size_t idx = (size_t)(mb + i) * DMODEL + n;
            out_pre[idx] = acc.c[s][i] + x_res[idx];
        }
    }
}

// ---------------------------------------------------------------------------
// LayerNorm over last dim (768) -> d_out
// ---------------------------------------------------------------------------
__global__ void k_layernorm(const float* __restrict__ out_pre,
                            const float* __restrict__ g,
                            const float* __restrict__ bta,
                            float* __restrict__ out) {
    const int D = DMODEL;
    int row = blockIdx.x, tid = threadIdx.x;      // 256 threads, 3 elems each
    __shared__ float s1[256], s2[256];
    size_t base = (size_t)row * D;
    float v0 = out_pre[base + tid];
    float v1 = out_pre[base + 256 + tid];
    float v2 = out_pre[base + 512 + tid];
    s1[tid] = v0 + v1 + v2;
    s2[tid] = v0 * v0 + v1 * v1 + v2 * v2;
    __syncthreads();
    for (int off = 128; off > 0; off >>= 1) {
        if (tid < off) { s1[tid] += s1[tid + off]; s2[tid] += s2[tid + off]; }
        __syncthreads();
    }
    float mu  = s1[0] / D;
    float var = s2[0] / D - mu * mu;
    float rs  = rsqrtf(var + 1e-5f);
    out[base + tid]       = (v0 - mu) * rs * g[tid]       + bta[tid];
    out[base + 256 + tid] = (v1 - mu) * rs * g[256 + tid] + bta[256 + tid];
    out[base + 512 + tid] = (v2 - mu) * rs * g[512 + tid] + bta[512 + tid];
}

// ---------------------------------------------------------------------------
// Host-side orchestration
// ---------------------------------------------------------------------------
extern "C" void kernel_launch(void* const* d_in, const int* in_sizes, int n_in,
                              void* d_out, int out_size, void* d_ws, size_t ws_size,
                              hipStream_t stream) {
    (void)in_sizes; (void)n_in; (void)out_size; (void)ws_size;

    const float* x     = (const float*)d_in[0];   // (2,1024,768)
    const float* W_in  = (const float*)d_in[1];   // (768,3072)
    const float* W_x   = (const float*)d_in[2];   // (1536,1664)
    const float* W_dt  = (const float*)d_in[3];   // (64,1536)
    const float* b_dt  = (const float*)d_in[4];   // (1536,)
    const float* A_log = (const float*)d_in[5];   // (1536,64)
    const float* Dp    = (const float*)d_in[6];   // (1536,)
    const float* W_out = (const float*)d_in[7];   // (1536,768)
    const float* ln_g  = (const float*)d_in[8];   // (768,)
    const float* ln_b  = (const float*)d_in[9];   // (768,)
    float* out = (float*)d_out;

    // Workspace carve-up (256B aligned)
    size_t off = 0;
    auto carve = [&](size_t bytes) -> char* {
        char* p = (char*)d_ws + off;
        off += (bytes + 255) & ~(size_t)255;
        return p;
    };
    _Float16* WinT  = (_Float16*)carve((size_t)(2*DINNER) * DMODEL * 2);  // 3072x768
    _Float16* WxT   = (_Float16*)carve((size_t)XDBL   * DINNER * 2);      // 1664x1536
    _Float16* WdtT  = (_Float16*)carve((size_t)DINNER * DSTATE * 2);      // 1536x64
    _Float16* WoutT = (_Float16*)carve((size_t)DMODEL * DINNER * 2);      // 768x1536
    _Float16* Xh    = (_Float16*)carve((size_t)MROWS * DMODEL * 2);
    _Float16* xs_h  = (_Float16*)carve((size_t)MROWS * DINNER * 2);
    float*    resS  = (float*)   carve((size_t)MROWS * DINNER * 4);
    float*    xdbl  = (float*)   carve((size_t)MROWS * XDBL   * 4);
    _Float16* dtinH = (_Float16*)carve((size_t)MROWS * DSTATE * 2);
    float*    delta = (float*)   carve((size_t)MROWS * DINNER * 4);
    _Float16* y_h   = (_Float16*)carve((size_t)MROWS * DINNER * 2);
    float*    outP  = (float*)   carve((size_t)MROWS * DMODEL * 4);

    const int TPB = 256;
    auto blocks = [](long n, int tpb) { return (int)((n + tpb - 1) / tpb); };

    // 1) pack weights (transposed f16) + activations (row-major f16)
    k_pack_t_f16<<<blocks((long)(2*DINNER)*DMODEL, TPB), TPB, 0, stream>>>(W_in,  WinT,  DMODEL, 2*DINNER);
    k_pack_t_f16<<<blocks((long)XDBL*DINNER,       TPB), TPB, 0, stream>>>(W_x,   WxT,   DINNER, XDBL);
    k_pack_t_f16<<<blocks((long)DINNER*DSTATE,     TPB), TPB, 0, stream>>>(W_dt,  WdtT,  DSTATE, DINNER);
    k_pack_t_f16<<<blocks((long)DMODEL*DINNER,     TPB), TPB, 0, stream>>>(W_out, WoutT, DINNER, DMODEL);
    k_pack_f16  <<<blocks((long)MROWS*DMODEL,      TPB), TPB, 0, stream>>>(x, Xh, MROWS*DMODEL);

    // 2) WMMA GEMM 1 + SiLU split      (64 x 96 tiles of 32x32, 8 waves/block)
    k_gemm_in   <<<(MROWS/32)*(2*DINNER/32)/8, TPB, 0, stream>>>(Xh, WinT, xs_h, resS);
    // 3) WMMA GEMM 2 -> x_dbl          (64 x 52 tiles)
    k_gemm_xdbl <<<(MROWS/32)*(XDBL/32)/8,     TPB, 0, stream>>>(xs_h, WxT, xdbl, dtinH);
    // 4) WMMA GEMM 3 + softplus clamp  (64 x 48 tiles)
    k_gemm_delta<<<(MROWS/32)*(DINNER/32)/8,   TPB, 0, stream>>>(dtinH, WdtT, b_dt, delta);
    // 5) selective scan (2*1536 channels / 256 per block)
    k_scan      <<<BATCH*(DINNER/256),         TPB, 0, stream>>>(xdbl, xs_h, delta, A_log, Dp, resS, y_h);
    // 6) WMMA GEMM 4 + residual        (64 x 24 tiles)
    k_gemm_out  <<<(MROWS/32)*(DMODEL/32)/8,   TPB, 0, stream>>>(y_h, WoutT, x, outP);
    // 7) LayerNorm -> d_out
    k_layernorm <<<MROWS,                      TPB, 0, stream>>>(outP, ln_g, ln_b, out);
}
